// LinearChainCrf_22591527977539
// MI455X (gfx1250) — compile-verified, matrix operand order checked by reading, not run on hardware
//
#include <hip/hip_runtime.h>

typedef __attribute__((ext_vector_type(16))) _Float16 v16h;
typedef __attribute__((ext_vector_type(8)))  float    v8f;

#define K_TAGS   128
#define BATCH    64
#define SEQ_T    512
#define NFEAT    8
#define NFEATURES 500000
#define TILE_F   64

// ---------------------------------------------------------------------------
// expm1(transitions) -> f16 table used as WMMA B-matrix source
// ---------------------------------------------------------------------------
__global__ __launch_bounds__(256) void crf_expm1_trans(
    const float* __restrict__ trans, _Float16* __restrict__ Ef)
{
  int i = blockIdx.x * 256 + threadIdx.x;
  if (i < K_TAGS * K_TAGS) Ef[i] = (_Float16)expm1f(trans[i]);
}

// ---------------------------------------------------------------------------
// Tiled transpose: sw[K][F] -> swT[F][K]  (both directions fully coalesced)
// ---------------------------------------------------------------------------
__global__ __launch_bounds__(256) void crf_transpose_sw(
    const float* __restrict__ sw, float* __restrict__ swT)
{
  __shared__ float tile[TILE_F][K_TAGS + 1];
  const int f0 = blockIdx.x * TILE_F;
  for (int idx = threadIdx.x; idx < K_TAGS * TILE_F; idx += 256) {
    int k = idx / TILE_F;
    int f = idx % TILE_F;
    float v = 0.f;
    if (f0 + f < NFEATURES) v = sw[(size_t)k * NFEATURES + (f0 + f)];
    tile[f][k] = v;                    // stride 129 dwords: conflict-free
  }
  __syncthreads();
  for (int idx = threadIdx.x; idx < K_TAGS * TILE_F; idx += 256) {
    int f = idx / K_TAGS;
    int k = idx % K_TAGS;
    if (f0 + f < NFEATURES) swT[(size_t)(f0 + f) * K_TAGS + k] = tile[f][k];
  }
}

// ---------------------------------------------------------------------------
// Emissions via transposed table: one wave per (b,t); float4 per lane -> each
// feature lookup is one coalesced 512B burst across the wave.
// ---------------------------------------------------------------------------
__global__ __launch_bounds__(256) void crf_emissions_transposed(
    const int* __restrict__ feats, const float* __restrict__ swT,
    float* __restrict__ em)
{
  int wid  = blockIdx.x * 8 + (threadIdx.x >> 5);
  int lane = threadIdx.x & 31;
  if (wid >= BATCH * SEQ_T) return;
  const int* f = feats + (size_t)wid * NFEAT;
  float4 acc = make_float4(0.f, 0.f, 0.f, 0.f);
#pragma unroll
  for (int q = 0; q < NFEAT; ++q) {
    const float4* col = (const float4*)(swT + (size_t)f[q] * K_TAGS);
    float4 v = col[lane];
    acc.x += v.x; acc.y += v.y; acc.z += v.z; acc.w += v.w;
  }
  ((float4*)(em + (size_t)wid * K_TAGS))[lane] = acc;
}

// Fallback: direct strided gather if the workspace can't hold swT.
__global__ __launch_bounds__(256) void crf_emissions_direct(
    const int* __restrict__ feats, const float* __restrict__ sw,
    float* __restrict__ em)
{
  int wid  = blockIdx.x * 8 + (threadIdx.x >> 5);
  int lane = threadIdx.x & 31;
  if (wid >= BATCH * SEQ_T) return;
  const int* f = feats + (size_t)wid * NFEAT;
  int fi[NFEAT];
#pragma unroll
  for (int q = 0; q < NFEAT; ++q) fi[q] = f[q];
  float acc[4] = {0.f, 0.f, 0.f, 0.f};
#pragma unroll
  for (int q = 0; q < NFEAT; ++q) {
#pragma unroll
    for (int c = 0; c < 4; ++c)
      acc[c] += sw[(size_t)(lane + 32 * c) * NFEATURES + fi[q]];
  }
#pragma unroll
  for (int c = 0; c < 4; ++c)
    em[(size_t)wid * K_TAGS + lane + 32 * c] = acc[c];
}

// ---------------------------------------------------------------------------
// Gold path score: one wave per batch, shuffle reduction over T.
// ---------------------------------------------------------------------------
__global__ __launch_bounds__(32) void crf_gold(
    const int* __restrict__ tags, const float* __restrict__ em,
    const float* __restrict__ trans, const float* __restrict__ startT,
    const float* __restrict__ endT, float* __restrict__ gold)
{
  int b = blockIdx.x;
  int lane = threadIdx.x;
  const int* tg = tags + (size_t)b * SEQ_T;
  float s = 0.f;
  for (int t = lane; t < SEQ_T; t += 32) {
    int k = tg[t];
    s += em[((size_t)b * SEQ_T + t) * K_TAGS + k];
    if (t + 1 < SEQ_T) s += trans[k * K_TAGS + tg[t + 1]];
  }
#pragma unroll
  for (int d = 1; d < 32; d <<= 1) s += __shfl_xor(s, d, 32);
  if (lane == 0) gold[b] = s + startT[tg[0]] + endT[tg[SEQ_T - 1]];
}

// ---------------------------------------------------------------------------
// Forward scan. One block = 16 batch rows, 8 waves = 8 tag-column tiles.
// alpha'[j] = M + log(S + (p @ expm1T)[j]) + em[t,j]; matmul on v_wmma f16.
// Two independent accumulators break the WMMA->WMMA RAW hazard chain;
// next-step emissions are prefetched a full iteration ahead.
// ---------------------------------------------------------------------------
__global__ __launch_bounds__(256) void crf_scan_kernel(
    const float* __restrict__ em,      // [B][T][K]
    const _Float16* __restrict__ Ef,   // [K][K] = expm1(trans)
    const float* __restrict__ startT,  // [K]
    const float* __restrict__ endT,    // [K]
    const float* __restrict__ gold,    // [B]
    float* __restrict__ out)           // [B]
{
  __shared__ float alpha[16][132];
  __shared__ float pbuf[16][132];
  __shared__ float Mrow[16];
  __shared__ float Srow[16];

  const int b0      = blockIdx.x * 16;
  const int wave    = threadIdx.x >> 5;      // owns tag tile [16w, 16w+16)
  const int lane    = threadIdx.x & 31;
  const int jbase   = wave * 16;
  const int halfsel = (lane >= 16) ? 1 : 0;
  const int l16     = lane & 15;

  // Preload B fragments (held in VGPRs for the entire T loop).
  // k(h,lane) per ISA 16-bit fragment layout.
  v16h bfrag[4];
#pragma unroll
  for (int c = 0; c < 4; ++c) {
#pragma unroll
    for (int h = 0; h < 16; ++h) {
      int k = 32 * c + h + ((h >= 8) ? 8 : 0) + (halfsel ? 8 : 0);
      bfrag[c][h] = Ef[k * K_TAGS + jbase + l16];
    }
  }

  // alpha0 = start + em[:,0]
  for (int idx = threadIdx.x; idx < 16 * K_TAGS; idx += 256) {
    int r = idx >> 7, j = idx & 127;
    alpha[r][j] = startT[j] + em[((size_t)(b0 + r) * SEQ_T) * K_TAGS + j];
  }
  __syncthreads();

  for (int t = 1; t < SEQ_T; ++t) {
    // Prefetch step t+1 emissions tile (16 rows x 128 f32 = 8KB) one
    // iteration ahead: 256 threads x 32B segments -> global_prefetch_b8.
    if (t + 1 < SEQ_T) {
      int pr  = threadIdx.x >> 4;          // 0..15 batch row
      int seg = (threadIdx.x & 15) * 8;    // 32B segment within the row
      __builtin_prefetch(
          &em[((size_t)(b0 + pr) * SEQ_T + (t + 1)) * K_TAGS + seg], 0, 3);
    }

    // Phase A: per-row max, p = exp(alpha - M), S = sum(p).
    // Wave w handles rows 2w (lanes 0-15) and 2w+1 (lanes 16-31); 8 cols/lane.
    {
      int r = 2 * wave + halfsel;
      int cb = l16 * 8;
      float a[8], m = -1e30f;
#pragma unroll
      for (int q = 0; q < 8; ++q) { a[q] = alpha[r][cb + q]; m = fmaxf(m, a[q]); }
#pragma unroll
      for (int d = 1; d < 16; d <<= 1) m = fmaxf(m, __shfl_xor(m, d, 32));
      float s = 0.f;
#pragma unroll
      for (int q = 0; q < 8; ++q) {
        float p = __expf(a[q] - m);
        pbuf[r][cb + q] = p;
        s += p;
      }
#pragma unroll
      for (int d = 1; d < 16; d <<= 1) s += __shfl_xor(s, d, 32);
      if (l16 == 0) { Mrow[r] = m; Srow[r] = s; }
    }
    __syncthreads();

    // Phase B: G = p @ expm1T for this wave's 16-column tile.
    // Even/odd K-chunks go to independent accumulators: adjacent WMMAs have
    // no D->A/B/C overlap, so no hazard NOPs between them.
    v8f acc0 = {}, acc1 = {};
#pragma unroll
    for (int c = 0; c < 4; c += 2) {
      v16h af0, af1;
#pragma unroll
      for (int h = 0; h < 16; ++h) {
        int kb = h + ((h >= 8) ? 8 : 0) + (halfsel ? 8 : 0);
        af0[h] = (_Float16)pbuf[l16][32 * c + kb];
        af1[h] = (_Float16)pbuf[l16][32 * (c + 1) + kb];
      }
      acc0 = __builtin_amdgcn_wmma_f32_16x16x32_f16(
          false, af0, false, bfrag[c],     (short)0, acc0, false, false);
      acc1 = __builtin_amdgcn_wmma_f32_16x16x32_f16(
          false, af1, false, bfrag[c + 1], (short)0, acc1, false, false);
    }

    // Phase C: alpha' = M + log(S + G) + em ; D row = d + 8*halfsel, col = l16
#pragma unroll
    for (int d = 0; d < 8; ++d) {
      int r = d + 8 * halfsel;
      int j = jbase + l16;
      float g = acc0[d] + acc1[d];
      float na = Mrow[r] + __logf(Srow[r] + g) +
                 em[((size_t)(b0 + r) * SEQ_T + t) * K_TAGS + j];
      alpha[r][j] = na;
    }
    __syncthreads();
  }

  // logZ = logsumexp(alpha + end); out = logZ - gold
  {
    int r = 2 * wave + halfsel;
    int cb = l16 * 8;
    float a[8], m = -1e30f;
#pragma unroll
    for (int q = 0; q < 8; ++q) {
      a[q] = alpha[r][cb + q] + endT[cb + q];
      m = fmaxf(m, a[q]);
    }
#pragma unroll
    for (int d = 1; d < 16; d <<= 1) m = fmaxf(m, __shfl_xor(m, d, 32));
    float s = 0.f;
#pragma unroll
    for (int q = 0; q < 8; ++q) s += __expf(a[q] - m);
#pragma unroll
    for (int d = 1; d < 16; d <<= 1) s += __shfl_xor(s, d, 32);
    if (l16 == 0) out[b0 + r] = (m + __logf(s)) - gold[b0 + r];
  }
}

// ---------------------------------------------------------------------------
extern "C" void kernel_launch(void* const* d_in, const int* in_sizes, int n_in,
                              void* d_out, int out_size, void* d_ws, size_t ws_size,
                              hipStream_t stream)
{
  (void)in_sizes; (void)n_in; (void)out_size;
  const int*   feats  = (const int*)  d_in[0];
  const int*   tags   = (const int*)  d_in[1];
  const float* sw     = (const float*)d_in[2];
  const float* trans  = (const float*)d_in[3];
  const float* startT = (const float*)d_in[4];
  const float* endT   = (const float*)d_in[5];
  float*       out    = (float*)d_out;

  char* ws = (char*)d_ws;
  float* em = (float*)ws;
  size_t off = (size_t)BATCH * SEQ_T * K_TAGS * sizeof(float);   // 16 MB
  _Float16* Ef = (_Float16*)(ws + off);
  off += (size_t)K_TAGS * K_TAGS * sizeof(_Float16);
  off = (off + 255) & ~(size_t)255;
  float* gold = (float*)(ws + off);
  off += BATCH * sizeof(float);
  off = (off + 255) & ~(size_t)255;
  const size_t swT_bytes = (size_t)NFEATURES * K_TAGS * sizeof(float); // 256 MB
  const bool use_transpose = (ws_size >= off + swT_bytes);

  crf_expm1_trans<<<(K_TAGS * K_TAGS + 255) / 256, 256, 0, stream>>>(trans, Ef);

  const int nblocks = (BATCH * SEQ_T) / 8;   // one wave per (b,t), 8 waves/block
  if (use_transpose) {
    float* swT = (float*)(ws + off);
    crf_transpose_sw<<<(NFEATURES + TILE_F - 1) / TILE_F, 256, 0, stream>>>(sw, swT);
    crf_emissions_transposed<<<nblocks, 256, 0, stream>>>(feats, swT, em);
  } else {
    crf_emissions_direct<<<nblocks, 256, 0, stream>>>(feats, sw, em);
  }

  crf_gold<<<BATCH, 32, 0, stream>>>(tags, em, trans, startT, endT, gold);
  crf_scan_kernel<<<BATCH / 16, 256, 0, stream>>>(em, Ef, startT, endT, gold, out);
}
